// GraphNetsConv_14731737825432
// MI455X (gfx1250) — matrix operand order repaired and checked by reading, not compiled, not run on hardware
//
#include <hip/hip_runtime.h>
#include <hip/hip_bf16.h>

// GraphNets conv layer for gfx1250 (MI455X), wave32 + WMMA (f16 in, f32 acc).
//
// Roofline: ~176 GFLOP total, ~360 MB HBM floor (~15 us @ 23.3 TB/s).
// f16 weights (1.8 MB) are L2-resident; M=32 edge tiles keep L2 weight
// re-reads at ~5 GB. Weights are pre-swizzled into per-lane WMMA fragment
// order so each B fragment is ONE contiguous 32B load per lane.
//
// d_out = [ new_x (N*D f32) | new_edge (E*D f32) ]; new_x region doubles as
// the scatter buffer `agg` (each node row is consumed only by the block that
// later overwrites it). d_ws = swizzled f16 weights (~1.8 MB).

typedef _Float16 half_t;
typedef half_t v16h __attribute__((ext_vector_type(16)));
typedef half_t v8h  __attribute__((ext_vector_type(8)));
typedef half_t v4h  __attribute__((ext_vector_type(4)));
typedef float  v8f  __attribute__((ext_vector_type(8)));

#define GN_D 256

// ---------------------------------------------------------------------------
// Fragment loaders (wave32, V_WMMA_F32_16X16X32_F16 layouts, ISA 7.12.2)
// ---------------------------------------------------------------------------

// A tile in LDS, row-major [16][LDA] f16, LDA multiple of 8 (16B-aligned rows).
// Lane L<16 holds row M=L, K = {kb+0..7, kb+16..23}; lanes 16..31: +8.
// Both 8-half groups are contiguous & 16B aligned -> two ds_load_b128.
template <int LDA>
__device__ __forceinline__ v16h load_a_frag(const half_t* __restrict__ a,
                                            int kBase, int lane) {
  const half_t* p =
      a + (size_t)(lane & 15) * LDA + kBase + ((lane >> 4) & 1) * 8;
  v8h lo = *(const v8h*)(p);
  v8h hi = *(const v8h*)(p + 16);
  return __builtin_shufflevector(lo, hi, 0, 1, 2, 3, 4, 5, 6, 7, 8, 9, 10, 11,
                                 12, 13, 14, 15);
}

// B fragments pre-swizzled in global memory: fragment (kt,nt), lane l owns
// 16 contiguous halves = W[kt*32 + (l>=16)*8 + (e&7) + (e>>3)*16][nt*16 + (l&15)].
// One aligned 32B load per lane, consecutive lanes contiguous (coalesced).
template <int NOUT>
__device__ __forceinline__ v16h load_b_frag(const half_t* __restrict__ Wf,
                                            int kt, int nt, int lane) {
  const v16h* p = (const v16h*)Wf;
  return p[((size_t)kt * (NOUT / 16) + nt) * 32 + lane];
}

__device__ __forceinline__ v8f wmma_f16(v16h a, v16h b, v8f c) {
  return __builtin_amdgcn_wmma_f32_16x16x32_f16(false, a, false, b, (short)0,
                                                c, false, false);
}

// ---------------------------------------------------------------------------
// Utility kernels
// ---------------------------------------------------------------------------

__global__ void zero_f32_kernel(float* __restrict__ p, int n) {
  for (int i = blockIdx.x * blockDim.x + threadIdx.x; i < n;
       i += gridDim.x * blockDim.x)
    p[i] = 0.0f;
}

// f32 [K][N] row-major -> f16 WMMA-fragment order (see load_b_frag).
template <int K, int N>
__global__ void swizzle_w_kernel(const float* __restrict__ src,
                                 half_t* __restrict__ dst) {
  const int total = K * N;
  for (int idx = blockIdx.x * blockDim.x + threadIdx.x; idx < total;
       idx += gridDim.x * blockDim.x) {
    int e = idx & 15;
    int lane = (idx >> 4) & 31;
    int t = idx >> 9;  // kt * (N/16) + nt
    int nt = t % (N / 16);
    int kt = t / (N / 16);
    int k = kt * 32 + ((lane >> 4) & 1) * 8 + (e & 7) + (e >> 3) * 16;
    int n = nt * 16 + (lane & 15);
    dst[idx] = (half_t)src[(size_t)k * N + n];
  }
}

// ---------------------------------------------------------------------------
// Edge processor, 32 edges per block (2 row-tiles share every B fragment):
//   h   = [x[i] | x[j] | edge_attr]            [32, 768]
//   hid = relu(h @ W1 + b1)                    [32, 512]
//   o   = hid @ W2 + b2                        [32, 256]
//   new_edge = edge_attr + LN(o); agg[j] += new_edge
// ---------------------------------------------------------------------------

__global__ __launch_bounds__(256) void edge_kernel(
    const float* __restrict__ x, const int* __restrict__ ei,
    const float* __restrict__ edge_attr, const half_t* __restrict__ W1f,
    const float* __restrict__ b1, const half_t* __restrict__ W2f,
    const float* __restrict__ b2, const float* __restrict__ g,
    const float* __restrict__ beta, float* __restrict__ new_edge,
    float* __restrict__ agg, int E) {
  constexpr int D = GN_D;    // 256
  constexpr int K1 = 3 * D;  // 768
  constexpr int H = 2 * D;   // 512
  constexpr int LDA = K1 + 8;
  constexpr int LDH = H + 8;

  // ob aliases aT: aT is dead once stage 1 finishes. ~83 KB total LDS
  // (gfx1250 WGP: up to 320 KB per workgroup).
  __shared__ union {
    half_t aT[32][LDA];  // 49.7 KB
    float ob[32][D + 4]; // 33.3 KB
  } u;
  __shared__ half_t hid[32][LDH];  // 33.3 KB
  __shared__ int iIdx[32], jIdx[32];
  __shared__ float mu_s[32], rs_s[32];

  const int tid = threadIdx.x;
  const int lane = tid & 31;
  const int wave = tid >> 5;
  const int eBase = blockIdx.x * 32;

  if (tid < 32) {
    iIdx[tid] = ei[eBase + tid];
    jIdx[tid] = ei[E + eBase + tid];
  }
  __syncthreads();

  // Stage A = [x[i] | x[j] | edge_attr] as f16, float4 granularity.
  for (int gq = tid; gq < 32 * (K1 / 4); gq += 256) {
    int r = gq / (K1 / 4);
    int k = (gq - r * (K1 / 4)) * 4;
    float4 v;
    if (k < D)          v = *(const float4*)&x[(size_t)iIdx[r] * D + k];
    else if (k < 2 * D) v = *(const float4*)&x[(size_t)jIdx[r] * D + (k - D)];
    else v = *(const float4*)&edge_attr[(size_t)(eBase + r) * D + (k - 2 * D)];
    v4h h;
    h[0] = (half_t)v.x; h[1] = (half_t)v.y;
    h[2] = (half_t)v.z; h[3] = (half_t)v.w;
    *(v4h*)&u.aT[r][k] = h;
  }
  __syncthreads();

  const int hi = (lane >> 4) & 1;
  const int lcol = lane & 15;

  // Stage 1: each wave owns 64 hidden cols (4 n-tiles) x 2 m-tiles.
  for (int nt2 = 0; nt2 < 4; ++nt2) {
    const int nt = wave * 4 + nt2;
    v8f acc0 = {}, acc1 = {};
    for (int kt = 0; kt < K1 / 32; ++kt) {
      v16h b = load_b_frag<H>(W1f, kt, nt, lane);
      v16h a0 = load_a_frag<LDA>(&u.aT[0][0], kt * 32, lane);
      v16h a1 = load_a_frag<LDA>(&u.aT[16][0], kt * 32, lane);
      acc0 = wmma_f16(a0, b, acc0);
      acc1 = wmma_f16(a1, b, acc1);
    }
    const int col = nt * 16 + lcol;
    const float bias = b1[col];
#pragma unroll
    for (int r = 0; r < 8; ++r) {
      float v0 = acc0[r] + bias;
      float v1 = acc1[r] + bias;
      hid[r + hi * 8][col] = (half_t)(v0 > 0.0f ? v0 : 0.0f);
      hid[16 + r + hi * 8][col] = (half_t)(v1 > 0.0f ? v1 : 0.0f);
    }
  }
  __syncthreads();  // aT dead from here; ob may be written.

  // Stage 2: each wave owns 32 output cols (2 n-tiles) x 2 m-tiles.
  for (int nt2 = 0; nt2 < 2; ++nt2) {
    const int nt = wave * 2 + nt2;
    v8f acc0 = {}, acc1 = {};
    for (int kt = 0; kt < H / 32; ++kt) {
      v16h b = load_b_frag<D>(W2f, kt, nt, lane);
      v16h a0 = load_a_frag<LDH>(&hid[0][0], kt * 32, lane);
      v16h a1 = load_a_frag<LDH>(&hid[16][0], kt * 32, lane);
      acc0 = wmma_f16(a0, b, acc0);
      acc1 = wmma_f16(a1, b, acc1);
    }
    const int col = nt * 16 + lcol;
    const float bias = b2[col];
#pragma unroll
    for (int r = 0; r < 8; ++r) {
      u.ob[r + hi * 8][col] = acc0[r] + bias;
      u.ob[16 + r + hi * 8][col] = acc1[r] + bias;
    }
  }
  __syncthreads();

  // Row-wise LayerNorm statistics (one thread per row).
  if (tid < 32) {
    float mu = 0.0f;
    for (int c = 0; c < D; ++c) mu += u.ob[tid][c];
    mu *= (1.0f / D);
    float var = 0.0f;
    for (int c = 0; c < D; ++c) {
      float d = u.ob[tid][c] - mu;
      var += d * d;
    }
    var *= (1.0f / D);
    mu_s[tid] = mu;
    rs_s[tid] = rsqrtf(var + 1e-5f);
  }
  __syncthreads();

  // Normalize, residual, write new_edge, scatter-add into agg[j].
  const int r = tid >> 3, c0 = tid & 7;  // 8 threads per row, 32 rows
  const float mu = mu_s[r], rs = rs_s[r];
  const size_t eoff = (size_t)(eBase + r) * D;
  const size_t joff = (size_t)jIdx[r] * D;
  for (int c = c0; c < D; c += 8) {
    float v = (u.ob[r][c] - mu) * rs * g[c] + beta[c];
    float ne = edge_attr[eoff + c] + v;
    new_edge[eoff + c] = ne;
    atomicAdd(&agg[joff + c], ne);
  }
}

// ---------------------------------------------------------------------------
// Node processor, 16 nodes per block (N=10000 -> 625 full tiles):
//   hn  = [x | agg]; hid = relu(hn @ W1 + b1); o = hid @ W2 + b2
//   new_x = x + LN(o)   (new_x aliases agg; rows are block-private)
// ---------------------------------------------------------------------------

__global__ __launch_bounds__(256) void node_kernel(
    const float* __restrict__ x, const float* __restrict__ agg,
    const half_t* __restrict__ W1f, const float* __restrict__ b1,
    const half_t* __restrict__ W2f, const float* __restrict__ b2,
    const float* __restrict__ g, const float* __restrict__ beta,
    float* __restrict__ new_x) {
  constexpr int D = GN_D;    // 256
  constexpr int K1 = 2 * D;  // 512
  constexpr int H = 2 * D;   // 512
  constexpr int LDA = K1 + 8;
  constexpr int LDH = H + 8;

  __shared__ union {
    half_t aT[16][LDA];
    float ob[16][D + 4];
  } u;
  __shared__ half_t hid[16][LDH];
  __shared__ float mu_s[16], rs_s[16];

  const int tid = threadIdx.x;
  const int lane = tid & 31;
  const int wave = tid >> 5;
  const int nBase0 = blockIdx.x * 16;

  for (int gq = tid; gq < 16 * (K1 / 4); gq += 256) {
    int r = gq / (K1 / 4);
    int k = (gq - r * (K1 / 4)) * 4;
    float4 v = (k < D) ? *(const float4*)&x[(size_t)(nBase0 + r) * D + k]
                       : *(const float4*)&agg[(size_t)(nBase0 + r) * D + (k - D)];
    v4h h;
    h[0] = (half_t)v.x; h[1] = (half_t)v.y;
    h[2] = (half_t)v.z; h[3] = (half_t)v.w;
    *(v4h*)&u.aT[r][k] = h;
  }
  __syncthreads();

  const int hi = (lane >> 4) & 1;
  const int lcol = lane & 15;

  for (int nt2 = 0; nt2 < 4; ++nt2) {
    const int nt = wave * 4 + nt2;
    v8f acc = {};
    for (int kt = 0; kt < K1 / 32; ++kt) {
      v16h b = load_b_frag<H>(W1f, kt, nt, lane);
      v16h a = load_a_frag<LDA>(&u.aT[0][0], kt * 32, lane);
      acc = wmma_f16(a, b, acc);
    }
    const int col = nt * 16 + lcol;
    const float bias = b1[col];
#pragma unroll
    for (int r = 0; r < 8; ++r) {
      float v = acc[r] + bias;
      hid[r + hi * 8][col] = (half_t)(v > 0.0f ? v : 0.0f);
    }
  }
  __syncthreads();

  for (int nt2 = 0; nt2 < 2; ++nt2) {
    const int nt = wave * 2 + nt2;
    v8f acc = {};
    for (int kt = 0; kt < H / 32; ++kt) {
      v16h b = load_b_frag<D>(W2f, kt, nt, lane);
      v16h a = load_a_frag<LDH>(&hid[0][0], kt * 32, lane);
      acc = wmma_f16(a, b, acc);
    }
    const int col = nt * 16 + lcol;
    const float bias = b2[col];
#pragma unroll
    for (int r = 0; r < 8; ++r) u.ob[r + hi * 8][col] = acc[r] + bias;
  }
  __syncthreads();

  if (tid < 16) {
    float mu = 0.0f;
    for (int c = 0; c < D; ++c) mu += u.ob[tid][c];
    mu *= (1.0f / D);
    float var = 0.0f;
    for (int c = 0; c < D; ++c) {
      float d = u.ob[tid][c] - mu;
      var += d * d;
    }
    var *= (1.0f / D);
    mu_s[tid] = mu;
    rs_s[tid] = rsqrtf(var + 1e-5f);
  }
  __syncthreads();

  const int r = tid >> 4, c0 = tid & 15;
  const float mu = mu_s[r], rs = rs_s[r];
  const size_t noff = (size_t)(nBase0 + r) * D;
  for (int c = c0; c < D; c += 16) {
    float v = (u.ob[r][c] - mu) * rs * g[c] + beta[c];
    new_x[noff + c] = x[noff + c] + v;
  }
}

// ---------------------------------------------------------------------------
// Launch
// ---------------------------------------------------------------------------

extern "C" void kernel_launch(void* const* d_in, const int* in_sizes, int n_in,
                              void* d_out, int out_size, void* d_ws,
                              size_t ws_size, hipStream_t stream) {
  (void)n_in; (void)out_size; (void)ws_size;
  constexpr int D = GN_D;

  const float* x    = (const float*)d_in[0];
  const int*   ei   = (const int*)d_in[1];    // [2, E] int32
  const float* ea   = (const float*)d_in[2];
  const float* e_w1 = (const float*)d_in[3];  // [768, 512]
  const float* e_b1 = (const float*)d_in[4];
  const float* e_w2 = (const float*)d_in[5];  // [512, 256]
  const float* e_b2 = (const float*)d_in[6];
  const float* e_g  = (const float*)d_in[7];
  const float* e_be = (const float*)d_in[8];
  const float* n_w1 = (const float*)d_in[9];  // [512, 512]
  const float* n_b1 = (const float*)d_in[10];
  const float* n_w2 = (const float*)d_in[11]; // [512, 256]
  const float* n_b2 = (const float*)d_in[12];
  const float* n_g  = (const float*)d_in[13];
  const float* n_be = (const float*)d_in[14];

  const int N = in_sizes[0] / D;  // 10000
  const int E = in_sizes[1] / 2;  // 160000

  float* new_x    = (float*)d_out;                  // [N, D] (doubles as agg)
  float* new_edge = (float*)d_out + (size_t)N * D;  // [E, D]
  float* agg      = new_x;

  // Swizzled f16 weights in workspace.
  char* ws = (char*)d_ws;
  half_t* ew1h = (half_t*)(ws);
  half_t* ew2h = (half_t*)(ws + 786432);
  half_t* nw1h = (half_t*)(ws + 786432 + 262144);
  half_t* nw2h = (half_t*)(ws + 786432 + 262144 + 524288);

  zero_f32_kernel<<<2048, 256, 0, stream>>>(agg, N * D);

  swizzle_w_kernel<768, 512><<<1024, 256, 0, stream>>>(e_w1, ew1h);
  swizzle_w_kernel<512, 256><<<1024, 256, 0, stream>>>(e_w2, ew2h);
  swizzle_w_kernel<512, 512><<<1024, 256, 0, stream>>>(n_w1, nw1h);
  swizzle_w_kernel<512, 256><<<1024, 256, 0, stream>>>(n_w2, nw2h);

  edge_kernel<<<E / 32, 256, 0, stream>>>(x, ei, ea, ew1h, e_b1, ew2h, e_b2,
                                          e_g, e_be, new_edge, agg, E);

  node_kernel<<<N / 16, 256, 0, stream>>>(x, agg, nw1h, n_b1, nw2h, n_b2, n_g,
                                          n_be, new_x);
}
